// LSTM_10393820857106
// MI455X (gfx1250) — compile-verified, matrix operand order checked by reading, not run on hardware
//
#include <hip/hip_runtime.h>

// ---------------------------------------------------------------------------
// LSTM on MI455X (gfx1250): bf16 WMMA (f32 accum) for both the big input
// projection GEMM and the 512 sequential recurrent steps.
//   B=64, T=512, F=1024, H=1024, 4H=4096
// Round 2: gemm_xu tile staging switched to gfx1250 async global->LDS DMA
// (global_load_async_to_lds_b128 + s_wait_asynccnt) with double-buffered LDS.
// ---------------------------------------------------------------------------

#define B_  64
#define T_  512
#define F_  1024
#define H_  1024
#define NG_ 4096          // 4*H
#define BT_ 32768         // B*T

typedef __attribute__((ext_vector_type(16))) __bf16 bf16x16;
typedef __attribute__((ext_vector_type(8)))  float  f32x8;

union Frag16 {            // one WMMA 16-bit A/B operand: 16 bf16 = 2 x 16B
    bf16x16 v;
    uint4   u[2];
};

__device__ __forceinline__ f32x8 wmma_bf16(const bf16x16& a, const bf16x16& b,
                                           const f32x8& c) {
    return __builtin_amdgcn_wmma_f32_16x16x32_bf16(
        /*neg_a=*/false, a, /*neg_b=*/false, b,
        /*c_mod=*/(short)0, c, /*reuse_a=*/false, /*reuse_b=*/false);
}

// gfx1250 async DMA: copy 16B from global to LDS, tracked by ASYNCcnt.
// VDST = VGPR with raw LDS byte address (low 32 bits of generic LDS pointer),
// VADDR = 64-bit global address VGPR pair, no SADDR ("off").
__device__ __forceinline__ void async_copy_b128(unsigned lds_byte_addr,
                                                const void* gaddr) {
    asm volatile("global_load_async_to_lds_b128 %0, %1, off"
                 :
                 : "v"(lds_byte_addr),
                   "v"((unsigned long long)(uintptr_t)gaddr)
                 : "memory");
}

__device__ __forceinline__ void wait_async0() {
    asm volatile("s_wait_asynccnt 0x0" ::: "memory");
}

__device__ __forceinline__ unsigned lds_addr32(const void* p) {
    return (unsigned)(uintptr_t)p;   // low 32 bits of generic ptr = LDS offset
}

// ---------------------------------------------------------------------------
// Packing kernels (one-time, bandwidth-trivial)
// ---------------------------------------------------------------------------

// Gate-concatenated TRANSPOSED bf16 weights: WT[n][k] = W_gate(n/H)[k][n%H].
// Transposed so WMMA B-fragments read contiguous 16B runs along k.
__global__ void pack_wT(const float* __restrict__ w0, const float* __restrict__ w1,
                        const float* __restrict__ w2, const float* __restrict__ w3,
                        __bf16* __restrict__ dst) {
    long idx = (long)blockIdx.x * blockDim.x + threadIdx.x;   // over 4*H*F
    if (idx >= (long)NG_ * F_) return;
    int n = (int)(idx / F_);
    int f = (int)(idx % F_);
    int g = n / H_;
    int h = n % H_;
    const float* src = (g == 0) ? w0 : (g == 1) ? w1 : (g == 2) ? w2 : w3;
    dst[idx] = (__bf16)src[(long)f * H_ + h];                  // coalesced write
}

__global__ void pack_bias(const float* __restrict__ b0, const float* __restrict__ b1,
                          const float* __restrict__ b2, const float* __restrict__ b3,
                          float* __restrict__ dst) {
    int n = blockIdx.x * blockDim.x + threadIdx.x;
    if (n >= NG_) return;
    int g = n / H_, h = n % H_;
    const float* src = (g == 0) ? b0 : (g == 1) ? b1 : (g == 2) ? b2 : b3;
    dst[n] = src[h];
}

__global__ void pack_x(const float* __restrict__ x, __bf16* __restrict__ xb) {
    long idx = (long)blockIdx.x * blockDim.x + threadIdx.x;   // over BT*F
    if (idx >= (long)BT_ * F_) return;
    xb[idx] = (__bf16)x[idx];
}

__global__ void init_state(__bf16* __restrict__ h0, float* __restrict__ c) {
    int idx = blockIdx.x * blockDim.x + threadIdx.x;          // over B*H
    if (idx >= B_ * H_) return;
    h0[idx] = (__bf16)0.0f;
    c[idx]  = 0.0f;
}

// ---------------------------------------------------------------------------
// Phase 1: xU[m][n] = sum_k Xb[m][k] * U[k][n] + b[n]
//   M=32768, N=4096, K=1024. Block tile 128x128, BK=32 (one WMMA K-slab).
//   8 waves, each owning a 32(M) x 64(N) sub-tile = 2x4 f32x8 accumulators.
//   Double-buffered LDS tiles filled by async global->LDS DMA; the async
//   copy of slab k+1 overlaps the 8 WMMAs of slab k. Sync per iteration:
//   each wave waits its own ASYNCcnt, then a workgroup barrier publishes
//   all waves' LDS writes.
//   LDS rows padded to 56 bf16 (112B): 16B-aligned b128 access, conflict-free
//   bank stride (28 DWORDs) for the 16-lane fragment reads.
// ---------------------------------------------------------------------------
__global__ __launch_bounds__(256) void gemm_xu(
        const __bf16* __restrict__ Xb,   // [BT][F]   row-major
        const __bf16* __restrict__ UT,   // [NG][F]   (transposed weights)
        const float*  __restrict__ bcat, // [NG]
        float*        __restrict__ xU) { // [BT][NG]
    __shared__ __bf16 As[2][128][56];
    __shared__ __bf16 Bs[2][128][56];

    const int tid  = threadIdx.x;
    const int lane = tid & 31;
    const int wave = tid >> 5;
    const int lm   = lane & 15;        // row/col within fragment
    const int kh   = lane >> 4;        // K half-select (ISA A/B layout)
    const int m0 = blockIdx.y * 128;
    const int n0 = blockIdx.x * 128;
    const int wm = (wave & 3) * 32;    // wave M offset inside block tile
    const int wn = (wave >> 2) * 64;   // wave N offset inside block tile

    // staging assignment: rows tid/4 and tid/4+64, 16B chunk tid%4
    const int srow = tid >> 2;
    const int scol = (tid & 3) * 8;    // bf16 elements

    f32x8 acc[2][4];
    for (int nf = 0; nf < 4; ++nf) {
        float bias = bcat[n0 + wn + nf * 16 + lm];
        for (int r = 0; r < 8; ++r) { acc[0][nf][r] = bias; acc[1][nf][r] = bias; }
    }

    // per-thread base pointers for staging
    const __bf16* gA0 = Xb + (size_t)(m0 + srow)      * F_ + scol;
    const __bf16* gA1 = Xb + (size_t)(m0 + srow + 64) * F_ + scol;
    const __bf16* gB0 = UT + (size_t)(n0 + srow)      * F_ + scol;
    const __bf16* gB1 = UT + (size_t)(n0 + srow + 64) * F_ + scol;

    auto issue_tile = [&](int stage, int kt) {
        const int kb = kt * 32;
        async_copy_b128(lds_addr32(&As[stage][srow][scol]),      gA0 + kb);
        async_copy_b128(lds_addr32(&As[stage][srow + 64][scol]), gA1 + kb);
        async_copy_b128(lds_addr32(&Bs[stage][srow][scol]),      gB0 + kb);
        async_copy_b128(lds_addr32(&Bs[stage][srow + 64][scol]), gB1 + kb);
    };

    issue_tile(0, 0);
    wait_async0();
    __syncthreads();

    for (int kt = 0; kt < 32; ++kt) {
        const int cur = kt & 1;
        if (kt < 31) issue_tile(cur ^ 1, kt + 1);   // DMA overlaps WMMAs below

        Frag16 af[2], bf[4];
        for (int mf = 0; mf < 2; ++mf) {
            int r = wm + mf * 16 + lm;
            af[mf].u[0] = *(const uint4*)&As[cur][r][kh * 8];
            af[mf].u[1] = *(const uint4*)&As[cur][r][16 + kh * 8];
        }
        for (int nf = 0; nf < 4; ++nf) {
            int r = wn + nf * 16 + lm;
            bf[nf].u[0] = *(const uint4*)&Bs[cur][r][kh * 8];
            bf[nf].u[1] = *(const uint4*)&Bs[cur][r][16 + kh * 8];
        }
        for (int mf = 0; mf < 2; ++mf)
            for (int nf = 0; nf < 4; ++nf)
                acc[mf][nf] = wmma_bf16(af[mf].v, bf[nf].v, acc[mf][nf]);

        wait_async0();       // own async stores to LDS stage cur^1 landed
        __syncthreads();     // publish across waves; also fences frag reads
    }

    // C/D layout: element r <-> (m = r + 8*kh, n = lm)
    for (int mf = 0; mf < 2; ++mf)
        for (int nf = 0; nf < 4; ++nf) {
            size_t mrow = (size_t)(m0 + wm + mf * 16 + 8 * kh);
            int    ncol = n0 + wn + nf * 16 + lm;
            for (int r = 0; r < 8; ++r)
                xU[(mrow + r) * NG_ + ncol] = acc[mf][nf][r];
        }
}

// ---------------------------------------------------------------------------
// Phase 2: one kernel launch per timestep (stream order = step sync).
//   Grid = 32 blocks: 4 batch-groups (16 rows) x 8 hidden-splits (128 units).
//   A block owns z-columns {g*H + hbase + j} for all 4 gates g -> the whole
//   gate nonlinearity + cell update is block-local after one barrier.
//   h is double-buffered across steps. A/B fragments read straight from
//   global (h: 128KB, V: 8MB -> L2 resident).
// ---------------------------------------------------------------------------
__global__ __launch_bounds__(256) void lstm_step(
        const float*  __restrict__ xU,   // [BT][NG]
        const __bf16* __restrict__ VT,   // [NG][H] transposed recurrent weights
        const __bf16* __restrict__ hin,  // [B][H]
        __bf16*       __restrict__ hout, // [B][H]
        float*        __restrict__ cst,  // [B][H]
        float*        __restrict__ out,  // [B][H] (final h)
        int t, int last) {
    __shared__ float zs[16][516];        // 16 batch rows x 512 z-cols (+pad)

    const int tid  = threadIdx.x;
    const int lane = tid & 31;
    const int wave = tid >> 5;
    const int lm   = lane & 15;
    const int kh   = lane >> 4;
    const int bg    = blockIdx.x & 3;
    const int hs    = blockIdx.x >> 2;
    const int b0    = bg * 16;
    const int hbase = hs * 128;
    const int g     = wave >> 1;         // gate handled by this wave
    const int halfw = wave & 1;          // which 64-col half of the 128 slice

    f32x8 acc[4];
    for (int nf = 0; nf < 4; ++nf)
        for (int r = 0; r < 8; ++r) acc[nf][r] = 0.0f;

    const __bf16* aRow = hin + (size_t)(b0 + lm) * H_;
    for (int kt = 0; kt < 32; ++kt) {
        const int kb = kt * 32 + kh * 8;
        Frag16 a;
        a.u[0] = *(const uint4*)&aRow[kb];
        a.u[1] = *(const uint4*)&aRow[kb + 16];
        for (int nf = 0; nf < 4; ++nf) {
            int n = g * H_ + hbase + halfw * 64 + nf * 16 + lm;
            const __bf16* bRow = VT + (size_t)n * H_;
            Frag16 b;
            b.u[0] = *(const uint4*)&bRow[kb];
            b.u[1] = *(const uint4*)&bRow[kb + 16];
            acc[nf] = wmma_bf16(a.v, b.v, acc[nf]);
        }
    }

    // spill z to LDS; local column c = g*128 + j (j = halfw*64 + nf*16 + lm)
    for (int nf = 0; nf < 4; ++nf)
        for (int r = 0; r < 8; ++r)
            zs[r + 8 * kh][wave * 64 + nf * 16 + lm] = acc[nf][r];
    __syncthreads();

    // gates + cell update: 16 rows x 128 hidden units = 2048 items
    for (int it = 0; it < 8; ++it) {
        int idx = tid + it * 256;
        int m = idx >> 7;
        int j = idx & 127;
        int b  = b0 + m;
        int hn = hbase + j;
        size_t xrow = ((size_t)b * T_ + t) * NG_;
        float zi = xU[xrow + 0 * H_ + hbase + j] + zs[m][0 * 128 + j];
        float zf = xU[xrow + 1 * H_ + hbase + j] + zs[m][1 * 128 + j];
        float zg = xU[xrow + 2 * H_ + hbase + j] + zs[m][2 * 128 + j];
        float zo = xU[xrow + 3 * H_ + hbase + j] + zs[m][3 * 128 + j];
        float iv = 1.0f / (1.0f + __expf(-zi));
        float fv = 1.0f / (1.0f + __expf(-zf));
        float gv = tanhf(zg);
        float ov = tanhf(zo);               // reference uses tanh for o-gate
        size_t sidx = (size_t)b * H_ + hn;
        float cn = fv * cst[sidx] + iv * gv;
        float hv = ov * tanhf(cn);
        cst[sidx]  = cn;
        hout[sidx] = (__bf16)hv;
        if (last) out[sidx] = hv;
    }
}

// ---------------------------------------------------------------------------
// Host driver
// ---------------------------------------------------------------------------
extern "C" void kernel_launch(void* const* d_in, const int* in_sizes, int n_in,
                              void* d_out, int out_size, void* d_ws, size_t ws_size,
                              hipStream_t stream) {
    // input order: x, then (U, V, b) per gate i,f,g,o
    const float* x  = (const float*)d_in[0];
    const float* Ui = (const float*)d_in[1];
    const float* Vi = (const float*)d_in[2];
    const float* bi = (const float*)d_in[3];
    const float* Uf = (const float*)d_in[4];
    const float* Vf = (const float*)d_in[5];
    const float* bf = (const float*)d_in[6];
    const float* Ug = (const float*)d_in[7];
    const float* Vg = (const float*)d_in[8];
    const float* bg = (const float*)d_in[9];
    const float* Uo = (const float*)d_in[10];
    const float* Vo = (const float*)d_in[11];
    const float* bo = (const float*)d_in[12];

    // workspace layout (all offsets 256B aligned)
    char* ws = (char*)d_ws;
    const size_t szUT = (size_t)NG_ * F_ * sizeof(__bf16);   //  8 MB
    const size_t szVT = (size_t)NG_ * H_ * sizeof(__bf16);   //  8 MB
    const size_t szB  = (size_t)NG_ * sizeof(float);         // 16 KB
    const size_t szXB = (size_t)BT_ * F_ * sizeof(__bf16);   // 64 MB
    const size_t szXU = (size_t)BT_ * NG_ * sizeof(float);   // 512 MB
    const size_t szH  = (size_t)B_ * H_ * sizeof(__bf16);    // 128 KB (x2)

    __bf16* UT   = (__bf16*)(ws);
    __bf16* VT   = (__bf16*)(ws + szUT);
    float*  bcat = (float*) (ws + szUT + szVT);
    __bf16* Xb   = (__bf16*)(ws + szUT + szVT + szB);
    float*  xU   = (float*) (ws + szUT + szVT + szB + szXB);
    __bf16* hbuf = (__bf16*)(ws + szUT + szVT + szB + szXB + szXU);
    float*  cst  = (float*) (ws + szUT + szVT + szB + szXB + szXU + 2 * szH);

    const int TPB = 256;

    pack_wT<<<(int)(((size_t)NG_ * F_ + TPB - 1) / TPB), TPB, 0, stream>>>(Ui, Uf, Ug, Uo, UT);
    pack_wT<<<(int)(((size_t)NG_ * H_ + TPB - 1) / TPB), TPB, 0, stream>>>(Vi, Vf, Vg, Vo, VT);
    pack_bias<<<(NG_ + TPB - 1) / TPB, TPB, 0, stream>>>(bi, bf, bg, bo, bcat);
    pack_x<<<(int)(((size_t)BT_ * F_ + TPB - 1) / TPB), TPB, 0, stream>>>(x, Xb);
    init_state<<<(B_ * H_ + TPB - 1) / TPB, TPB, 0, stream>>>(hbuf, cst);

    dim3 gGemm(NG_ / 128, BT_ / 128);   // (32, 256)
    gemm_xu<<<gGemm, TPB, 0, stream>>>(Xb, UT, bcat, xU);

    const size_t HN = (size_t)B_ * H_;
    for (int t = 0; t < T_; ++t) {
        lstm_step<<<32, TPB, 0, stream>>>(
            xU, VT,
            hbuf + (size_t)(t & 1) * HN,
            hbuf + (size_t)((t + 1) & 1) * HN,
            cst, (float*)d_out, t, t == T_ - 1 ? 1 : 0);
    }
}
